// HybridQwen3LaCTBranch_70377334112905
// MI455X (gfx1250) — matrix operand order repaired and checked by reading, not compiled, not run on hardware
//
#include <hip/hip_runtime.h>
#include <hip/hip_bf16.h>
#include <stdint.h>

#define DEV __device__ __forceinline__

typedef __bf16 bf16_t;
typedef __attribute__((ext_vector_type(8)))  __bf16 bf16x8;
typedef __attribute__((ext_vector_type(16))) __bf16 bf16x16;
typedef __attribute__((ext_vector_type(8)))  float  f32x8;
typedef __attribute__((ext_vector_type(4)))  unsigned int u32x4;
typedef __attribute__((ext_vector_type(8)))  int i32x8;
typedef __attribute__((ext_vector_type(4)))  int i32x4;

DEV int lane_id() { return threadIdx.x & 31; }
DEV int wave_id() { return threadIdx.x >> 5; }

DEV float sigmf(float x) { return 1.0f / (1.0f + __expf(-x)); }
DEV float siluf(float x) { return x * sigmf(x); }

DEV f32x8 zero8() { return (f32x8){0.f,0.f,0.f,0.f,0.f,0.f,0.f,0.f}; }

DEV f32x8 wmma_bf16(bf16x16 a, bf16x16 b, f32x8 c) {
  return __builtin_amdgcn_wmma_f32_16x16x32_bf16(false, a, false, b, (short)0, c, false, false);
}

// ---- WMMA fragment helpers (CDNA5 wave32 layouts, CDNA5 ISA) ----
// A 16x32 bf16: lanes 0-15 hold M=lane, K 0..15; lanes 16-31 K 16..31
DEV bf16x16 load_a_frag(const bf16_t* base, int m0, int k0, int ld) {
  int l = lane_id();
  const bf16_t* p = base + (m0 + (l & 15)) * ld + k0 + ((l >> 4) << 4);
  bf16x8 lo = *(const bf16x8*)p;
  bf16x8 hi = *(const bf16x8*)(p + 8);
  bf16x16 r;
#pragma unroll
  for (int i = 0; i < 8; ++i) { r[i] = lo[i]; r[i + 8] = hi[i]; }
  return r;
}
// B 32x16 bf16 from a row-major [n][k] source: lanes 0-15 K 0..15, lanes 16-31 K 16..31
DEV bf16x16 load_b_frag_row(const bf16_t* bt, int n0, int k0, int ld) {
  int l = lane_id();
  const bf16_t* p = bt + (n0 + (l & 15)) * ld + k0 + ((l >> 4) << 4);
  bf16x8 lo = *(const bf16x8*)p;
  bf16x8 hi = *(const bf16x8*)(p + 8);
  bf16x16 r;
#pragma unroll
  for (int i = 0; i < 8; ++i) { r[i] = lo[i]; r[i + 8] = hi[i]; }
  return r;
}
// B 32x16 bf16 from a row-major [k][n] source (strided gather)
DEV bf16x16 load_b_frag_col(const bf16_t* bm, int k0, int n0, int ld) {
  int l = lane_id();
  int n = n0 + (l & 15);
  int kk = k0 + ((l >> 4) << 4);
  bf16x16 r;
#pragma unroll
  for (int i = 0; i < 16; ++i) r[i] = bm[(kk + i) * ld + n];
  return r;
}
// C 16x16 f32: element (m = m0 + r + 8*(lane>=16), n = n0 + lane&15)
DEV void store_c_bf16(bf16_t* base, int m0, int n0, int ld, f32x8 c) {
  int l = lane_id(); int n = n0 + (l & 15); int mb = m0 + ((l >> 4) << 3);
#pragma unroll
  for (int r = 0; r < 8; ++r) base[(mb + r) * ld + n] = (bf16_t)c[r];
}
DEV void store_c_bf16_T(bf16_t* base, int m0, int n0, int ld, f32x8 c) { // dest[n][m]
  int l = lane_id(); int n = n0 + (l & 15); int mb = m0 + ((l >> 4) << 3);
#pragma unroll
  for (int r = 0; r < 8; ++r) base[n * ld + mb + r] = (bf16_t)c[r];
}

// ---- async global -> LDS (gfx1250, tracked by ASYNCcnt) ----
DEV void async_copy_b128(uint32_t lds_off, const void* g) {
  asm volatile("global_load_async_to_lds_b128 %0, %1, off"
               :: "v"(lds_off), "v"((uint64_t)(uintptr_t)g) : "memory");
}
DEV void wait_async0() { asm volatile("s_wait_asynccnt 0" ::: "memory"); }

// ---- TDM: DMA a contiguous 128x128 bf16 tile global -> LDS (TENSORcnt) ----
// D# per CDNA5 ISA: group0 {count, lds_addr, global_addr, type=2},
// group1 {data_size=2B, tensor_dim0/1=128, tile_dim0/1=128, dim0_stride=128}.
// This toolchain exposes the 6-arg builtin: (g0, g1, g2, g3, g4, cpol).
DEV void tdm_load_tile_128x128_bf16(uint32_t lds_off, const void* gaddr) {
  uint64_t ga = (uint64_t)(uintptr_t)gaddr;
  u32x4 g0;
  g0[0] = 1u;                                               // count=1 (valid user D#)
  g0[1] = lds_off;                                          // lds_addr (bytes)
  g0[2] = (uint32_t)ga;                                     // global_addr[31:0]
  g0[3] = (uint32_t)((ga >> 32) & 0x1FFFFFFu) | (2u << 30); // addr[56:32] | type=2
  i32x8 g1;
  g1[0] = (int)(1u << 16);        // workgroup_mask=0, data_size=1 (2 bytes)
  g1[1] = (int)(128u << 16);      // tensor_dim0[15:0]=128 (bits 63:48)
  g1[2] = (int)(128u << 16);      // tensor_dim0 hi=0 | tensor_dim1[15:0]=128
  g1[3] = (int)(128u << 16);      // tensor_dim1 hi=0 | tile_dim0=128
  g1[4] = 128;                    // tile_dim1=128, tile_dim2=0
  g1[5] = 128;                    // tensor_dim0_stride[31:0]=128
  g1[6] = 0;
  g1[7] = 0;
  i32x4 z4 = {0, 0, 0, 0};
  i32x8 z8 = {0, 0, 0, 0, 0, 0, 0, 0};
  __builtin_amdgcn_tensor_load_to_lds(g0, g1, z4, z4, z8, 0);
}

// ============================================================================
// Kernel 1: QKV GEMM  Y(16384x6144) = X(16384x2048) * W(6144x2048)^T, f32 out
// Double-buffered LDS tiles; B fragments preloaded so WMMAs issue back-to-back.
// ============================================================================
__global__ __launch_bounds__(256) void qkv_gemm_kernel(const float* __restrict__ X,
                                                       const float* __restrict__ W,
                                                       float* __restrict__ Y) {
  __shared__ bf16_t At[2][128 * 32];
  __shared__ bf16_t Bt[2][128 * 32];
  const int m0 = blockIdx.x * 128;
  const int n0 = blockIdx.y * 128;
  const int tid = threadIdx.x, w = wave_id(), l = lane_id();
  const int n_ = l & 15, hl = (l >> 4) << 3;
  f32x8 acc[8];
#pragma unroll
  for (int i = 0; i < 8; ++i) acc[i] = zero8();
  const int row = tid >> 1, col = (tid & 1) * 16;
  const float* xa0 = X + (long)(m0 + row) * 2048 + col;
  const float* xb0 = W + (long)(n0 + row) * 2048 + col;

  auto fill = [&](int buf, int k) {
    const float* xa = xa0 + k;
    const float* xb = xb0 + k;
#pragma unroll
    for (int j = 0; j < 16; j += 4) {
      float4 va = *(const float4*)(xa + j);
      float4 vb = *(const float4*)(xb + j);
      bf16_t* ap = &At[buf][row * 32 + col + j];
      bf16_t* bp = &Bt[buf][row * 32 + col + j];
      ap[0] = (bf16_t)va.x; ap[1] = (bf16_t)va.y; ap[2] = (bf16_t)va.z; ap[3] = (bf16_t)va.w;
      bp[0] = (bf16_t)vb.x; bp[1] = (bf16_t)vb.y; bp[2] = (bf16_t)vb.z; bp[3] = (bf16_t)vb.w;
    }
  };

  fill(0, 0);
  __syncthreads();
  int buf = 0;
  for (int k = 0; k < 2048; k += 32) {
    if (k + 32 < 2048) {
      __builtin_prefetch(xa0 + k + 64, 0, 0); // pull the tile after next into L2
      __builtin_prefetch(xb0 + k + 64, 0, 0);
      fill(buf ^ 1, k + 32);
    }
    bf16x16 af = load_a_frag(At[buf], w * 16, 0, 32);
    bf16x16 bfr[8];
#pragma unroll
    for (int i = 0; i < 8; ++i) bfr[i] = load_b_frag_row(Bt[buf], i * 16, 0, 32);
#pragma unroll
    for (int i = 0; i < 8; ++i) acc[i] = wmma_bf16(af, bfr[i], acc[i]);
    __syncthreads();
    buf ^= 1;
  }
#pragma unroll
  for (int i = 0; i < 8; ++i)
#pragma unroll
    for (int r = 0; r < 8; ++r)
      Y[(long)(m0 + w * 16 + hl + r) * 6144 + n0 + i * 16 + n_] = acc[i][r];
}

// ============================================================================
// Kernel 2: lr / momentum projections (64 dot products of length 2048 / token)
// ============================================================================
__global__ __launch_bounds__(256) void lrmom_kernel(const float* __restrict__ X,
                                                    const float* __restrict__ lr_w,
                                                    const float* __restrict__ lr_b,
                                                    const float* __restrict__ mom_w,
                                                    const float* __restrict__ mom_b,
                                                    float* __restrict__ lro,
                                                    float* __restrict__ momo) {
  __shared__ float xs[2048];
  const int row = blockIdx.x;
  const int b = row >> 13, s = row & 8191;
  const int tid = threadIdx.x, w = wave_id(), l = lane_id();
  for (int i = tid; i < 2048; i += 256) xs[i] = X[(long)row * 2048 + i];
  __syncthreads();
  const float BASE_LR_INV = -6.90725524f; // log(expm1(0.001))
  for (int oo = w; oo < 64; oo += 8) {
    const float* wr = (oo < 48) ? (lr_w + (long)oo * 2048) : (mom_w + (long)(oo - 48) * 2048);
    float p = 0.f;
    for (int i = l; i < 2048; i += 32) p += xs[i] * wr[i];
#pragma unroll
    for (int mk = 16; mk; mk >>= 1) p += __shfl_xor(p, mk, 32);
    if (l == 0) {
      if (oo < 48) {
        int h = oo / 3, j = oo - h * 3;
        float x = p + lr_b[oo] + BASE_LR_INV;
        float sp = (x > 20.f) ? x : log1pf(__expf(x));
        lro[(((long)(b * 16 + h)) * 8192 + s) * 3 + j] = sp;
      } else {
        int h = oo - 48;
        momo[((long)(b * 16 + h)) * 8192 + s] = sigmf(p + mom_b[h]);
      }
    }
  }
}

// ============================================================================
// Kernel 3: fused RMSNorm + scale/offset + silu + per-head l2n + RoPE -> bf16
// ============================================================================
__global__ __launch_bounds__(256) void fuse_kernel(const float* __restrict__ qkv,
                                                   const int* __restrict__ pos_ids,
                                                   const float* __restrict__ qn_w,
                                                   const float* __restrict__ kn_w,
                                                   const float* __restrict__ qk_scale,
                                                   const float* __restrict__ qk_off,
                                                   bf16_t* __restrict__ fq,
                                                   bf16_t* __restrict__ fk,
                                                   bf16_t* __restrict__ fv) {
  __shared__ float qs[2048], ks[2048], vs[2048];
  __shared__ float red[256];
  __shared__ float hq[16], hk[16];
  const int row = blockIdx.x;
  const int b = row >> 13, s = row & 8191;
  const int tid = threadIdx.x;
  const float* base = qkv + (long)row * 6144;
  float pq = 0.f, pk = 0.f;
  for (int i = tid; i < 2048; i += 256) {
    float q = base[i], k = base[2048 + i], v = base[4096 + i];
    qs[i] = q; ks[i] = k; vs[i] = v;
    pq += q * q; pk += k * k;
  }
  red[tid] = pq; __syncthreads();
  for (int st = 128; st; st >>= 1) { if (tid < st) red[tid] += red[tid + st]; __syncthreads(); }
  float invq = rsqrtf(red[0] * (1.f / 2048.f) + 1e-6f); __syncthreads();
  red[tid] = pk; __syncthreads();
  for (int st = 128; st; st >>= 1) { if (tid < st) red[tid] += red[tid + st]; __syncthreads(); }
  float invk = rsqrtf(red[0] * (1.f / 2048.f) + 1e-6f); __syncthreads();
  for (int i = tid; i < 2048; i += 256) {
    float q1 = qn_w[i] * (qs[i] * invq) * qk_scale[2 * i + 0] + qk_off[2 * i + 0];
    float k1 = kn_w[i] * (ks[i] * invk) * qk_scale[2 * i + 1] + qk_off[2 * i + 1];
    qs[i] = siluf(q1); ks[i] = siluf(k1); vs[i] = siluf(vs[i]);
  }
  __syncthreads();
  const int ht = tid >> 4, lt = tid & 15;
  float sq = 0.f, sk = 0.f;
#pragma unroll
  for (int j = 0; j < 8; ++j) {
    int i = ht * 128 + lt * 8 + j;
    sq += qs[i] * qs[i]; sk += ks[i] * ks[i];
  }
  red[tid] = sq; __syncthreads();
  if (lt == 0) { float t = 0; for (int j = 0; j < 16; ++j) t += red[ht * 16 + j]; hq[ht] = rsqrtf(t + 1e-6f); }
  __syncthreads();
  red[tid] = sk; __syncthreads();
  if (lt == 0) { float t = 0; for (int j = 0; j < 16; ++j) t += red[ht * 16 + j]; hk[ht] = rsqrtf(t + 1e-6f); }
  __syncthreads();
  const float pos = (float)pos_ids[row];
  const float LOGB = 13.815510558f; // ln(1e6)
  long obase = ((long)(b * 16 + ht) * 8192 + s) * 128;
#pragma unroll
  for (int j = 0; j < 8; ++j) {
    int d = lt * 8 + j, i = ht * 128 + d, fi = d & 63;
    float ang = pos * __expf(-(float)(2 * fi) * (1.0f / 128.0f) * LOGB);
    float sn, cs;
    __sincosf(ang, &sn, &cs);
    float qv = qs[i] * hq[ht], kv = ks[i] * hk[ht];
    float qr = (d < 64) ? -qs[ht * 128 + d + 64] * hq[ht] : qs[ht * 128 + d - 64] * hq[ht];
    float kr = (d < 64) ? -ks[ht * 128 + d + 64] * hk[ht] : ks[ht * 128 + d - 64] * hk[ht];
    fq[obase + d] = (bf16_t)(qv * cs + qr * sn);
    fk[obase + d] = (bf16_t)(kv * cs + kr * sn);
    fv[obase + d] = (bf16_t)vs[i];
  }
}

// ============================================================================
// Kernel 4: block-causal LaCT swiglu scan. One workgroup per (b,h) stream.
// ============================================================================
DEV void update_state(float* wstm, float* mstm, const f32x8* dwa, bf16_t* wb, float mbar) {
  const int w = wave_id(), l = lane_id();
  const int n_ = l & 15, hl = (l >> 4) << 3;
  f32x8 wraw[8];
  float ss[8];
#pragma unroll
  for (int r = 0; r < 8; ++r) ss[r] = 0.f;
#pragma unroll
  for (int i = 0; i < 8; ++i) {
#pragma unroll
    for (int r = 0; r < 8; ++r) {
      int idx = (w * 16 + hl + r) * 128 + i * 16 + n_;
      float m2 = mbar * mstm[idx] + dwa[i][r];
      mstm[idx] = m2;
      float wr2 = wstm[idx] + m2;
      wraw[i][r] = wr2;
      ss[r] += wr2 * wr2;
    }
  }
#pragma unroll
  for (int mk = 1; mk < 16; mk <<= 1)
#pragma unroll
    for (int r = 0; r < 8; ++r) ss[r] += __shfl_xor(ss[r], mk, 32);
#pragma unroll
  for (int r = 0; r < 8; ++r) ss[r] = rsqrtf(ss[r] + 1e-6f);
#pragma unroll
  for (int i = 0; i < 8; ++i)
#pragma unroll
    for (int r = 0; r < 8; ++r) {
      int idx = (w * 16 + hl + r) * 128 + i * 16 + n_;
      float v = wraw[i][r] * ss[r];
      wstm[idx] = v;
      wb[idx] = (bf16_t)v;
    }
}

__global__ __launch_bounds__(256) void lact_kernel(const bf16_t* __restrict__ fq,
                                                   const bf16_t* __restrict__ fk,
                                                   const bf16_t* __restrict__ fv,
                                                   const float* __restrict__ lrv,
                                                   const float* __restrict__ momv,
                                                   const float* __restrict__ w0in,
                                                   const float* __restrict__ w1in,
                                                   const float* __restrict__ w2in,
                                                   float* __restrict__ wst,
                                                   float* __restrict__ mst,
                                                   float* __restrict__ fwx) {
  extern __shared__ char smem_raw[];
  bf16_t* w0b = (bf16_t*)smem_raw;      // 3 weight operands + 6-buffer tile pool
  bf16_t* w1b = w0b + 16384;
  bf16_t* w2b = w1b + 16384;
  bf16_t* P0 = w2b + 16384;
  bf16_t* P1 = P0 + 16384;
  bf16_t* P2 = P1 + 16384;
  bf16_t* P3 = P2 + 16384;
  bf16_t* P4 = P3 + 16384;
  bf16_t* P5 = P4 + 16384;
  float* red = (float*)(P5 + 16384);

  const int bh = blockIdx.x, h = bh & 15;
  const int tid = threadIdx.x, w = wave_id(), l = lane_id();
  const int n_ = l & 15, hl = (l >> 4) << 3;

  float* ws0 = wst + (long)bh * 3 * 16384; float* ws1 = ws0 + 16384; float* ws2 = ws1 + 16384;
  float* ms0 = mst + (long)bh * 3 * 16384; float* ms1 = ms0 + 16384; float* ms2 = ms1 + 16384;

  for (int i = tid; i < 16384; i += 256) {
    float a0 = w0in[h * 16384 + i], a1 = w1in[h * 16384 + i], a2 = w2in[h * 16384 + i];
    ws0[i] = a0; ws1[i] = a1; ws2[i] = a2;
    w0b[i] = (bf16_t)a0; w1b[i] = (bf16_t)a1; w2b[i] = (bf16_t)a2;
    ms0[i] = 0.f; ms1[i] = 0.f; ms2[i] = 0.f;
  }
  __threadfence();
  __syncthreads();

  const uint32_t L0 = (uint32_t)(uintptr_t)P0;
  const uint32_t L1 = (uint32_t)(uintptr_t)P1;
  const uint32_t L2 = (uint32_t)(uintptr_t)P2;

  for (int c = 0; c < 4; ++c) {
    const long sbase = (long)bh * 8192 + c * 2048;
    float pm = 0.f;
    for (int i = tid; i < 2048; i += 256) pm += momv[sbase + i];
    red[tid] = pm; __syncthreads();
    for (int st = 128; st; st >>= 1) { if (tid < st) red[tid] += red[tid + st]; __syncthreads(); }
    const float mbar = red[0] * (1.0f / 2048.0f);
    __syncthreads();

    f32x8 dw0a[8], dw1a[8], dw2a[8];
#pragma unroll
    for (int i = 0; i < 8; ++i) { dw0a[i] = zero8(); dw1a[i] = zero8(); dw2a[i] = zero8(); }

    for (int rt = 0; rt < 16; ++rt) {
      __syncthreads(); // all prior-tile reads of P0..P2 finished before refill
      const long row0 = sbase + rt * 128;
      const long gb = row0 * 128;
      // V tile via Tensor Data Mover (TENSORcnt), Q/K tiles via async-to-LDS.
      if (w == 0) tdm_load_tile_128x128_bf16(L2, fv + gb);
#pragma unroll
      for (int i = 0; i < 8; ++i) {
        int lin = (i * 256 + tid) * 8;
        async_copy_b128(L0 + lin * 2, fq + gb + lin);
        async_copy_b128(L1 + lin * 2, fk + gb + lin);
      }
      wait_async0();
      if (w == 0) __builtin_amdgcn_s_wait_tensorcnt(0);
      __syncthreads();

      { // hidq = silu(Q w0^T) * (Q w2^T)  -> P3
        f32x8 a1[8], a2[8];
#pragma unroll
        for (int i = 0; i < 8; ++i) { a1[i] = zero8(); a2[i] = zero8(); }
#pragma unroll
        for (int kk = 0; kk < 4; ++kk) {
          bf16x16 af = load_a_frag(P0, w * 16, kk * 32, 128);
#pragma unroll
          for (int i = 0; i < 8; ++i) {
            a1[i] = wmma_bf16(af, load_b_frag_row(w0b, i * 16, kk * 32, 128), a1[i]);
            a2[i] = wmma_bf16(af, load_b_frag_row(w2b, i * 16, kk * 32, 128), a2[i]);
          }
        }
#pragma unroll
        for (int i = 0; i < 8; ++i) {
          f32x8 hv;
#pragma unroll
          for (int r = 0; r < 8; ++r) hv[r] = siluf(a1[i][r]) * a2[i][r];
          store_c_bf16(P3, w * 16, i * 16, 128, hv);
        }
      }
      __syncthreads();

      { // o = hidq * w1^T -> fwx (pre-update weights)
        f32x8 oc[8];
#pragma unroll
        for (int i = 0; i < 8; ++i) oc[i] = zero8();
#pragma unroll
        for (int kk = 0; kk < 4; ++kk) {
          bf16x16 af = load_a_frag(P3, w * 16, kk * 32, 128);
#pragma unroll
          for (int i = 0; i < 8; ++i)
            oc[i] = wmma_bf16(af, load_b_frag_row(w1b, i * 16, kk * 32, 128), oc[i]);
        }
#pragma unroll
        for (int i = 0; i < 8; ++i)
#pragma unroll
          for (int r = 0; r < 8; ++r)
            fwx[(row0 + w * 16 + hl + r) * 128 + i * 16 + n_] = oc[i][r];
      }
      __syncthreads();

      { // g -> P3, u -> P0, hid^T -> P4
        f32x8 gg[8], uu[8];
#pragma unroll
        for (int i = 0; i < 8; ++i) { gg[i] = zero8(); uu[i] = zero8(); }
#pragma unroll
        for (int kk = 0; kk < 4; ++kk) {
          bf16x16 af = load_a_frag(P1, w * 16, kk * 32, 128);
#pragma unroll
          for (int i = 0; i < 8; ++i) {
            gg[i] = wmma_bf16(af, load_b_frag_row(w0b, i * 16, kk * 32, 128), gg[i]);
            uu[i] = wmma_bf16(af, load_b_frag_row(w2b, i * 16, kk * 32, 128), uu[i]);
          }
        }
#pragma unroll
        for (int i = 0; i < 8; ++i) {
          f32x8 hid;
#pragma unroll
          for (int r = 0; r < 8; ++r) hid[r] = siluf(gg[i][r]) * uu[i][r];
          store_c_bf16(P3, w * 16, i * 16, 128, gg[i]);
          store_c_bf16(P0, w * 16, i * 16, 128, uu[i]);
          store_c_bf16_T(P4, w * 16, i * 16, 128, hid);
        }
      }
      __syncthreads();

      { // dh = V * w1 ; then dgb^T -> P5, du^T -> P3
        f32x8 dhf[8];
#pragma unroll
        for (int i = 0; i < 8; ++i) dhf[i] = zero8();
#pragma unroll
        for (int kk = 0; kk < 4; ++kk) {
          bf16x16 af = load_a_frag(P2, w * 16, kk * 32, 128);
#pragma unroll
          for (int i = 0; i < 8; ++i)
            dhf[i] = wmma_bf16(af, load_b_frag_col(w1b, kk * 32, i * 16, 128), dhf[i]);
        }
        f32x8 dgbf[8], duf[8];
#pragma unroll
        for (int i = 0; i < 8; ++i)
#pragma unroll
          for (int r = 0; r < 8; ++r) {
            int m = w * 16 + hl + r, nn = i * 16 + n_;
            float gv = (float)P3[m * 128 + nn];
            float uv = (float)P0[m * 128 + nn];
            float sg = sigmf(gv);
            float dh = dhf[i][r];
            duf[i][r] = dh * gv * sg;                              // dh * silu(g)
            dgbf[i][r] = dh * uv * sg * (1.0f + gv * (1.0f - sg)); // dh*u*sg*(1+g(1-sg))
          }
        __syncthreads();
#pragma unroll
        for (int i = 0; i < 8; ++i) {
          store_c_bf16_T(P5, w * 16, i * 16, 128, dgbf[i]);
          store_c_bf16_T(P3, w * 16, i * 16, 128, duf[i]);
        }
      }
      __syncthreads();

      // (K*lr1)^T -> P0 ; dw0 += dgb^T x (K*lr1)
      for (int idx = tid; idx < 16384; idx += 256) {
        int cl = idx >> 7, d = idx & 127;
        P0[d * 128 + cl] = (bf16_t)((float)P1[idx] * lrv[(row0 + cl) * 3 + 0]);
      }
      __syncthreads();
#pragma unroll
      for (int kk = 0; kk < 4; ++kk) {
        bf16x16 af = load_a_frag(P5, w * 16, kk * 32, 128);
#pragma unroll
        for (int i = 0; i < 8; ++i)
          dw0a[i] = wmma_bf16(af, load_b_frag_row(P0, i * 16, kk * 32, 128), dw0a[i]);
      }
      __syncthreads();

      // (K*lr3)^T -> P0 ; dw2 += du^T x (K*lr3)
      for (int idx = tid; idx < 16384; idx += 256) {
        int cl = idx >> 7, d = idx & 127;
        P0[d * 128 + cl] = (bf16_t)((float)P1[idx] * lrv[(row0 + cl) * 3 + 2]);
      }
      __syncthreads();
#pragma unroll
      for (int kk = 0; kk < 4; ++kk) {
        bf16x16 af = load_a_frag(P3, w * 16, kk * 32, 128);
#pragma unroll
        for (int i = 0; i < 8; ++i)
          dw2a[i] = wmma_bf16(af, load_b_frag_row(P0, i * 16, kk * 32, 128), dw2a[i]);
      }
      __syncthreads();

      // (V*lr2)^T -> P0 ; dw1 += (V*lr2)^T x hid
      for (int idx = tid; idx < 16384; idx += 256) {
        int cl = idx >> 7, o = idx & 127;
        P0[o * 128 + cl] = (bf16_t)((float)P2[idx] * lrv[(row0 + cl) * 3 + 1]);
      }
      __syncthreads();
#pragma unroll
      for (int kk = 0; kk < 4; ++kk) {
        bf16x16 af = load_a_frag(P0, w * 16, kk * 32, 128);
#pragma unroll
        for (int i = 0; i < 8; ++i)
          dw1a[i] = wmma_bf16(af, load_b_frag_row(P4, i * 16, kk * 32, 128), dw1a[i]);
      }
    } // row tiles

    __syncthreads();
    update_state(ws0, ms0, dw0a, w0b, mbar);
    update_state(ws1, ms1, dw1a, w1b, mbar);
    update_state(ws2, ms2, dw2a, w2b, mbar);
    __threadfence();
    __syncthreads();
  } // chunks
}

// ============================================================================
// Kernel 5: final per-head RMS norm + (BH,S,D) -> (B,S,H) transpose
// ============================================================================
__global__ __launch_bounds__(256) void outnorm_kernel(const float* __restrict__ fwx,
                                                      const float* __restrict__ tn_w,
                                                      float* __restrict__ out) {
  __shared__ float red[256];
  __shared__ float hn[16];
  const int row = blockIdx.x;
  const int b = row >> 13, s = row & 8191;
  const int tid = threadIdx.x, ht = tid >> 4, lt = tid & 15;
  long ib = ((long)(b * 16 + ht) * 8192 + s) * 128;
  float x[8]; float ss = 0.f;
#pragma unroll
  for (int j = 0; j < 8; ++j) { x[j] = fwx[ib + lt * 8 + j]; ss += x[j] * x[j]; }
  red[tid] = ss; __syncthreads();
  if (lt == 0) { float t = 0; for (int j = 0; j < 16; ++j) t += red[ht * 16 + j]; hn[ht] = rsqrtf(t * (1.f / 128.f) + 1e-6f); }
  __syncthreads();
#pragma unroll
  for (int j = 0; j < 8; ++j) {
    int d = lt * 8 + j;
    out[(long)row * 2048 + ht * 128 + d] = tn_w[d] * x[j] * hn[ht];
  }
}

// ============================================================================
extern "C" void kernel_launch(void* const* d_in, const int* in_sizes, int n_in,
                              void* d_out, int out_size, void* d_ws, size_t ws_size,
                              hipStream_t stream) {
  (void)in_sizes; (void)n_in; (void)out_size; (void)ws_size;
  const float* hidden = (const float*)d_in[0];
  const int* pos = (const int*)d_in[1];
  const float* qkv_w = (const float*)d_in[2];
  const float* qnw = (const float*)d_in[3];
  const float* knw = (const float*)d_in[4];
  const float* qks = (const float*)d_in[5];
  const float* qko = (const float*)d_in[6];
  const float* w0 = (const float*)d_in[7];
  const float* w1 = (const float*)d_in[8];
  const float* w2 = (const float*)d_in[9];
  const float* lr_w = (const float*)d_in[10];
  const float* lr_b = (const float*)d_in[11];
  const float* mom_w = (const float*)d_in[12];
  const float* mom_b = (const float*)d_in[13];
  const float* tnw = (const float*)d_in[14];
  float* out = (float*)d_out;

  char* ws = (char*)d_ws;
  size_t off = 0;
  auto alloc = [&](size_t bytes) -> char* {
    char* p = ws + off;
    off += (bytes + 255) & ~(size_t)255;
    return p;
  };
  float* qkv = (float*)alloc((size_t)16384 * 6144 * 4);
  bf16_t* fqw = (bf16_t*)alloc((size_t)32 * 8192 * 128 * 2);
  bf16_t* fkw = (bf16_t*)alloc((size_t)32 * 8192 * 128 * 2);
  bf16_t* fvw = (bf16_t*)alloc((size_t)32 * 8192 * 128 * 2);
  float* lrw = (float*)alloc((size_t)32 * 8192 * 3 * 4);
  float* momw = (float*)alloc((size_t)32 * 8192 * 4);
  float* fwx = (float*)alloc((size_t)32 * 8192 * 128 * 4);
  float* wst = (float*)alloc((size_t)3 * 32 * 16384 * 4);
  float* mst = (float*)alloc((size_t)3 * 32 * 16384 * 4);

  qkv_gemm_kernel<<<dim3(128, 48), 256, 0, stream>>>(hidden, qkv_w, qkv);
  lrmom_kernel<<<16384, 256, 0, stream>>>(hidden, lr_w, lr_b, mom_w, mom_b, lrw, momw);
  fuse_kernel<<<16384, 256, 0, stream>>>(qkv, pos, qnw, knw, qks, qko, fqw, fkw, fvw);

  const int lact_lds = 9 * 16384 * 2 + 256 * 4; // 9 x 32KB bf16 buffers + reduction
  (void)hipFuncSetAttribute((const void*)lact_kernel,
                            hipFuncAttributeMaxDynamicSharedMemorySize, lact_lds);
  lact_kernel<<<32, 256, lact_lds, stream>>>(fqw, fkw, fvw, lrw, momw, w0, w1, w2,
                                             wst, mst, fwx);

  outnorm_kernel<<<16384, 256, 0, stream>>>(fwx, tnw, out);
}
